// Model_72696616452483
// MI455X (gfx1250) — compile-verified, matrix operand order checked by reading, not run on hardware
//
#include <hip/hip_runtime.h>
#include <hip/hip_bf16.h>

#define N_NODES 10000
#define N_EDGES 320000
#define D_INK   128
#define D_HIDK  64
#define D_OUTK  16
#define NVEC    (N_NODES * D_HIDK)   // 640000

typedef __bf16 bf16_t;
typedef __attribute__((ext_vector_type(16))) __bf16 v16bf;
typedef __attribute__((ext_vector_type(8)))  float  v8f;

static __device__ __forceinline__ v8f wmma_bf16(v16bf a, v16bf b, v8f c) {
  return __builtin_amdgcn_wmma_f32_16x16x32_bf16(false, a, false, b, (short)0, c,
                                                 false, false);
}

union V16 { v16bf v; uint4 q[2]; };

// one 32B contiguous chunk -> v16bf (two b128 loads, no packing movs)
static __device__ __forceinline__ v16bf load_v16(const bf16_t* p) {
  V16 u;
  u.q[0] = ((const uint4*)p)[0];
  u.q[1] = ((const uint4*)p)[1];
  return u.v;
}
// two 16B chunks (A operand: K-halves 16 apart) -> v16bf
static __device__ __forceinline__ v16bf load_v16_split(const bf16_t* lo,
                                                       const bf16_t* hi) {
  V16 u;
  u.q[0] = *(const uint4*)lo;
  u.q[1] = *(const uint4*)hi;
  return u.v;
}

// convert 8 contiguous f32 -> 8 bf16, single 16B store (ds_store_b128 in LDS)
static __device__ __forceinline__ void cvt_store8(bf16_t* dst,
                                                  const float* __restrict__ src) {
  float4 f0 = ((const float4*)src)[0];
  float4 f1 = ((const float4*)src)[1];
  union { bf16_t h[8]; uint4 q; } pk;
  pk.h[0] = (bf16_t)f0.x; pk.h[1] = (bf16_t)f0.y;
  pk.h[2] = (bf16_t)f0.z; pk.h[3] = (bf16_t)f0.w;
  pk.h[4] = (bf16_t)f1.x; pk.h[5] = (bf16_t)f1.y;
  pk.h[6] = (bf16_t)f1.z; pk.h[7] = (bf16_t)f1.w;
  *(uint4*)dst = pk.q;
}
static __device__ __forceinline__ void cvt_store8_relu(bf16_t* dst,
                                                       const float* __restrict__ src) {
  float4 f0 = ((const float4*)src)[0];
  float4 f1 = ((const float4*)src)[1];
  union { bf16_t h[8]; uint4 q; } pk;
  pk.h[0] = (bf16_t)fmaxf(f0.x, 0.f); pk.h[1] = (bf16_t)fmaxf(f0.y, 0.f);
  pk.h[2] = (bf16_t)fmaxf(f0.z, 0.f); pk.h[3] = (bf16_t)fmaxf(f0.w, 0.f);
  pk.h[4] = (bf16_t)fmaxf(f1.x, 0.f); pk.h[5] = (bf16_t)fmaxf(f1.y, 0.f);
  pk.h[6] = (bf16_t)fmaxf(f1.z, 0.f); pk.h[7] = (bf16_t)fmaxf(f1.w, 0.f);
  *(uint4*)dst = pk.q;
}

// ---------------------------------------------------------------------------
// Theta = (I + Om)^-1 (I - Om), Om = 0.5(B - B^T). Gauss-Jordan, 64 threads
// (pivots of I+skew are >= 1, no pivoting needed). Emits Theta pre-swizzled
// into WMMA B-operand register order: [wtile(4)][ks(2)][lane(32)][16 bf16].
// ---------------------------------------------------------------------------
__global__ void __launch_bounds__(64)
theta_kernel(const float* __restrict__ cayB, bf16_t* __restrict__ thSw) {
  __shared__ float A[64][65];
  __shared__ float M[64][65];
  const int r = threadIdx.x;
  for (int c = 0; c < 64; ++c) {
    float om  = 0.5f * (cayB[r * 64 + c] - cayB[c * 64 + r]);
    float eye = (r == c) ? 1.f : 0.f;
    A[r][c] = eye + om;
    M[r][c] = eye - om;
  }
  __syncthreads();
  for (int p = 0; p < 64; ++p) {
    if (r == p) {
      float inv = 1.f / A[p][p];
      for (int c = 0; c < 64; ++c) { A[p][c] *= inv; M[p][c] *= inv; }
    }
    __syncthreads();
    if (r != p) {
      float f = A[r][p];
      for (int c = 0; c < 64; ++c) {
        A[r][c] -= f * A[p][c];
        M[r][c] -= f * M[p][c];
      }
    }
    __syncthreads();
  }
  // M now holds Theta[h][k]; swizzle: B[k][h]=Theta[h][k] in per-lane order
  for (int t2 = 0; t2 < 4; ++t2) {
    int q   = r * 4 + t2;              // ((wt*2+ks)*32 + lane)
    int ln  = q & 31;
    int ksw = q >> 5;
    int wt  = ksw >> 1, ks = ksw & 1;
    int h   = wt * 16 + (ln & 15);
    int kb  = ks * 32 + ((ln >> 4) << 3);
    for (int j = 0; j < 16; ++j) {
      int k = kb + (j < 8 ? j : 8 + j);   // j>=8 -> kb+16+(j-8)
      thSw[q * 16 + j] = (bf16_t)M[h][k];
    }
  }
}

// ---------------------------------------------------------------------------
// One-time swizzle of enc_W / dec_W into WMMA B-operand register order.
// enc: [wt(4)][ks(4)][lane(32)][16] = 8192; dec: [ks(2)][lane(32)][16] = 1024
// ---------------------------------------------------------------------------
__global__ void __launch_bounds__(256)
wswizzle_kernel(const float* __restrict__ encW, const float* __restrict__ decW,
                bf16_t* __restrict__ encWsw, bf16_t* __restrict__ decWsw) {
  int i = blockIdx.x * 256 + threadIdx.x;
  if (i < 8192) {
    int j = i & 15, q = i >> 4;
    int ln = q & 31, ksw = q >> 5;     // ksw = wt*4+ks
    int wt = ksw >> 2, ks = ksw & 3;
    int h  = wt * 16 + (ln & 15);
    int kb = ks * 32 + ((ln >> 4) << 3);
    int k  = kb + (j < 8 ? j : 8 + j);
    encWsw[i] = (bf16_t)encW[k * D_HIDK + h];
  } else if (i < 8192 + 1024) {
    int ii = i - 8192;
    int j = ii & 15, q = ii >> 4;
    int ln = q & 31, ks = q >> 5;
    int o  = ln & 15;
    int kb = ks * 32 + ((ln >> 4) << 3);
    int k  = kb + (j < 8 ? j : 8 + j);
    decWsw[ii] = (bf16_t)decW[k * D_OUTK + o];
  }
}

// ---------------------------------------------------------------------------
// Encoder: b[node][h] = x[node][:] @ enc_W[:, h] + enc_b[h]
// 16-node tile / block, 4 waves = 4 h-tiles, K=128 (4 bf16 WMMA steps).
// ---------------------------------------------------------------------------
__global__ void __launch_bounds__(128)
enc_kernel(const float* __restrict__ x, const bf16_t* __restrict__ Wsw,
           const float* __restrict__ bias, float* __restrict__ bout) {
  __shared__ __align__(16) bf16_t xl[16 * D_INK];   // [m][k]
  __shared__ float bl[D_HIDK];
  const int tid = threadIdx.x, tile = blockIdx.x;
  const float* xt = x + (size_t)tile * 16 * D_INK;  // tile rows contiguous
  cvt_store8(&xl[tid * 16],     &xt[tid * 16]);
  cvt_store8(&xl[tid * 16 + 8], &xt[tid * 16 + 8]);
  if (tid < D_HIDK) bl[tid] = bias[tid];
  __syncthreads();

  const int lane = tid & 31, wave = tid >> 5;
  const int m = lane & 15, kboff = (lane >> 4) * 8;
  const int h = wave * 16 + (lane & 15);
  v8f acc = {};
  for (int ks = 0; ks < 4; ++ks) {
    const int k0 = ks * 32 + kboff;
    v16bf a  = load_v16_split(&xl[m * D_INK + k0], &xl[m * D_INK + k0 + 16]);
    v16bf bb = load_v16(&Wsw[((wave * 4 + ks) * 32 + lane) * 16]);
    acc = wmma_bf16(a, bb, acc);
  }
  const int mtop = (lane >> 4) * 8;
#pragma unroll
  for (int v = 0; v < 8; ++v) {
    int node = tile * 16 + mtop + v;
    bout[node * D_HIDK + h] = acc[v] + bl[h];
  }
}

// ---------------------------------------------------------------------------
// t = 2*relu(u) - u + b   (optionally snapshot u into Anderson X history)
// ---------------------------------------------------------------------------
__global__ void __launch_bounds__(256)
tmap_kernel(const float* __restrict__ u, const float* __restrict__ b,
            float* __restrict__ t, float* __restrict__ xcopy) {
  int i = blockIdx.x * blockDim.x + threadIdx.x;
  if (i < NVEC) {
    float uv = u[i];
    float r  = uv > 0.f ? uv : 0.f;
    t[i] = 2.f * r - uv + b[i];
    if (xcopy) xcopy[i] = uv;
  }
}

// ---------------------------------------------------------------------------
// SpMM scatter: s[dst][c] += w * t[src][c]. One (edge,channel)/thread; edge
// scalars are wave-uniform -> readfirstlane => scalar (SMEM) loads; t/s
// accesses are 128B-coalesced per wave; f32 atomics resolve in L2.
// ---------------------------------------------------------------------------
__global__ void __launch_bounds__(256)
spmm_kernel(const float* __restrict__ t, const int* __restrict__ src,
            const int* __restrict__ dst, const float* __restrict__ w,
            float* __restrict__ s) {
  long gid = (long)blockIdx.x * 256 + threadIdx.x;
  int e = (int)(gid >> 6);                      // constant within a wave32
  int c = (int)(gid & 63);
  int eu = __builtin_amdgcn_readfirstlane(e);
  if (eu < N_EDGES) {
    int sn = src[eu], dn = dst[eu];
    float val = w[eu] * t[sn * D_HIDK + c];
    atomicAdd(&s[dn * D_HIDK + c], val);
  }
}

// ---------------------------------------------------------------------------
// Fused channel mix + PR update: out = b + 0.5*(s @ Theta^T)
// (== b + 0.5*Theta@spmm in reference layout). 2 WMMA k-steps, 4 h-tiles.
// ---------------------------------------------------------------------------
__global__ void __launch_bounds__(128)
mix_kernel(const float* __restrict__ s, const bf16_t* __restrict__ thSw,
           const float* __restrict__ b, float* __restrict__ out) {
  __shared__ __align__(16) bf16_t sl[16 * D_HIDK];  // [m][k]
  const int tid = threadIdx.x, tile = blockIdx.x;
  const float* st = s + (size_t)tile * 16 * D_HIDK; // tile contiguous
  cvt_store8(&sl[tid * 8], &st[tid * 8]);           // 1024 elems, 8/thread
  __syncthreads();

  const int lane = tid & 31, wave = tid >> 5;
  const int m = lane & 15, kboff = (lane >> 4) * 8;
  const int h = wave * 16 + (lane & 15);
  v8f acc = {};
  for (int ks = 0; ks < 2; ++ks) {
    const int k0 = ks * 32 + kboff;
    v16bf a  = load_v16_split(&sl[m * D_HIDK + k0], &sl[m * D_HIDK + k0 + 16]);
    v16bf bb = load_v16(&thSw[((wave * 2 + ks) * 32 + lane) * 16]);
    acc = wmma_bf16(a, bb, acc);
  }
  const int mtop = (lane >> 4) * 8;
#pragma unroll
  for (int v = 0; v < 8; ++v) {
    int node = tile * 16 + mtop + v;
    out[node * D_HIDK + h] = b[node * D_HIDK + h] + 0.5f * acc[v];
  }
}

// ---------------------------------------------------------------------------
// Anderson Gram: 15 upper-triangular entries of (F-X)(F-X)^T.
// Wave32 shuffle reduction + f32 atomics.
// ---------------------------------------------------------------------------
__global__ void __launch_bounds__(256)
dots_kernel(const float* __restrict__ X, const float* __restrict__ F,
            float* __restrict__ gg) {
  float acc[15];
#pragma unroll
  for (int c = 0; c < 15; ++c) acc[c] = 0.f;
  for (int i = blockIdx.x * blockDim.x + threadIdx.x; i < NVEC;
       i += gridDim.x * blockDim.x) {
    float g[5];
#pragma unroll
    for (int mm = 0; mm < 5; ++mm) g[mm] = F[mm * NVEC + i] - X[mm * NVEC + i];
    int c = 0;
#pragma unroll
    for (int a = 0; a < 5; ++a)
#pragma unroll
      for (int bj = a; bj < 5; ++bj) acc[c++] += g[a] * g[bj];
  }
  const int lane = threadIdx.x & 31;
#pragma unroll
  for (int c = 0; c < 15; ++c) {
    float v = acc[c];
    for (int off = 16; off > 0; off >>= 1) v += __shfl_down(v, off, 32);
    if (lane == 0) atomicAdd(&gg[c], v);
  }
}

// ---------------------------------------------------------------------------
// 5x5 regularized solve, a = normalize(GG^-1 * ones). Single scalar thread.
// ---------------------------------------------------------------------------
__global__ void solve_kernel(const float* __restrict__ gg, float* __restrict__ a) {
  if (threadIdx.x == 0 && blockIdx.x == 0) {
    float G[5][5], r[5];
    int c = 0;
    for (int i = 0; i < 5; ++i)
      for (int j = i; j < 5; ++j) { G[i][j] = gg[c]; G[j][i] = gg[c]; ++c; }
    for (int i = 0; i < 5; ++i) { G[i][i] += 1e-4f; r[i] = 1.f; }
    for (int p = 0; p < 5; ++p) {
      float inv = 1.f / G[p][p];
      for (int j = 0; j < 5; ++j) G[p][j] *= inv;
      r[p] *= inv;
      for (int q = 0; q < 5; ++q) if (q != p) {
        float f = G[q][p];
        for (int j = 0; j < 5; ++j) G[q][j] -= f * G[p][j];
        r[q] -= f * r[p];
      }
    }
    float invs = 1.f / (r[0] + r[1] + r[2] + r[3] + r[4]);
    for (int i = 0; i < 5; ++i) a[i] = r[i] * invs;
  }
}

// ---------------------------------------------------------------------------
// u_new = sum_i a[i]*F[i]; also writes Anderson X slot.
// ---------------------------------------------------------------------------
__global__ void __launch_bounds__(256)
combine_kernel(const float* __restrict__ F, const float* __restrict__ a,
               float* __restrict__ ucur, float* __restrict__ xslot) {
  int i = blockIdx.x * blockDim.x + threadIdx.x;
  if (i < NVEC) {
    float v = 0.f;
#pragma unroll
    for (int mm = 0; mm < 5; ++mm) v += a[mm] * F[mm * NVEC + i];
    ucur[i] = v;
    xslot[i] = v;
  }
}

// ---------------------------------------------------------------------------
// Decoder: out[node][o] = relu(u[node][:]) @ dec_W[:, o]. 1 wave / 16 nodes.
// ---------------------------------------------------------------------------
__global__ void __launch_bounds__(32)
dec_kernel(const float* __restrict__ u, const bf16_t* __restrict__ Wsw,
           float* __restrict__ out) {
  __shared__ __align__(16) bf16_t zl[16 * D_HIDK];  // [m][k] = relu(u)
  const int lane = threadIdx.x, tile = blockIdx.x;
  const float* ut = u + (size_t)tile * 16 * D_HIDK;
#pragma unroll
  for (int t2 = 0; t2 < 4; ++t2)                    // 1024 elems, 32/lane
    cvt_store8_relu(&zl[(t2 * 32 + lane) * 8], &ut[(t2 * 32 + lane) * 8]);
  __syncthreads();

  const int m = lane & 15, kboff = (lane >> 4) * 8;
  const int o = lane & 15;
  v8f acc = {};
  for (int ks = 0; ks < 2; ++ks) {
    const int k0 = ks * 32 + kboff;
    v16bf a  = load_v16_split(&zl[m * D_HIDK + k0], &zl[m * D_HIDK + k0 + 16]);
    v16bf bb = load_v16(&Wsw[(ks * 32 + lane) * 16]);
    acc = wmma_bf16(a, bb, acc);
  }
  const int mtop = (lane >> 4) * 8;
#pragma unroll
  for (int v = 0; v < 8; ++v) {
    int node = tile * 16 + mtop + v;
    out[node * D_OUTK + o] = acc[v];
  }
}

// ---------------------------------------------------------------------------
extern "C" void kernel_launch(void* const* d_in, const int* in_sizes, int n_in,
                              void* d_out, int out_size, void* d_ws, size_t ws_size,
                              hipStream_t stream) {
  (void)in_sizes; (void)n_in; (void)out_size; (void)ws_size;
  const float* x    = (const float*)d_in[0];
  const int*   ei   = (const int*)d_in[1];
  const float* ew   = (const float*)d_in[2];
  const float* encW = (const float*)d_in[3];
  const float* encb = (const float*)d_in[4];
  const float* cayB = (const float*)d_in[5];
  const float* decW = (const float*)d_in[6];
  const int* src = ei;
  const int* dst = ei + N_EDGES;

  char* p = (char*)d_ws;
  auto alloc = [&](size_t bytes) {
    char* q = p;
    p += (bytes + 255) & ~(size_t)255;
    return q;
  };
  float*  bvec   = (float*)alloc((size_t)NVEC * 4);
  float*  tvec   = (float*)alloc((size_t)NVEC * 4);
  float*  svec   = (float*)alloc((size_t)NVEC * 4);
  float*  ucur   = (float*)alloc((size_t)NVEC * 4);
  float*  X      = (float*)alloc((size_t)5 * NVEC * 4);
  float*  F      = (float*)alloc((size_t)5 * NVEC * 4);
  bf16_t* thSw   = (bf16_t*)alloc(4096 * 2);
  bf16_t* encWsw = (bf16_t*)alloc(8192 * 2);
  bf16_t* decWsw = (bf16_t*)alloc(1024 * 2);
  float*  gg     = (float*)alloc(16 * 4);
  float*  avec   = (float*)alloc(8 * 4);

  theta_kernel<<<1, 64, 0, stream>>>(cayB, thSw);
  wswizzle_kernel<<<36, 256, 0, stream>>>(encW, decW, encWsw, decWsw);
  enc_kernel<<<N_NODES / 16, 128, 0, stream>>>(x, encWsw, encb, bvec);

  // 5 plain PR sweeps filling Anderson history (X[i]=u_i, F[i]=g(u_i))
  const float* u = bvec;
  for (int i = 0; i < 5; ++i) {
    tmap_kernel<<<NVEC / 256, 256, 0, stream>>>(u, bvec, tvec,
                                                X + (size_t)i * NVEC);
    hipMemsetAsync(svec, 0, (size_t)NVEC * 4, stream);
    spmm_kernel<<<(N_EDGES * 64) / 256, 256, 0, stream>>>(tvec, src, dst, ew,
                                                          svec);
    mix_kernel<<<N_NODES / 16, 128, 0, stream>>>(svec, thSw, bvec,
                                                 F + (size_t)i * NVEC);
    u = F + (size_t)i * NVEC;
  }

  // 5 Anderson-accelerated steps (last g() is dead — output only needs u_new)
  for (int k = 0; k < 5; ++k) {
    hipMemsetAsync(gg, 0, 16 * 4, stream);
    dots_kernel<<<640, 256, 0, stream>>>(X, F, gg);
    solve_kernel<<<1, 32, 0, stream>>>(gg, avec);
    combine_kernel<<<NVEC / 256, 256, 0, stream>>>(F, avec, ucur,
                                                   X + (size_t)k * NVEC);
    if (k < 4) {
      tmap_kernel<<<NVEC / 256, 256, 0, stream>>>(ucur, bvec, tvec, nullptr);
      hipMemsetAsync(svec, 0, (size_t)NVEC * 4, stream);
      spmm_kernel<<<(N_EDGES * 64) / 256, 256, 0, stream>>>(tvec, src, dst, ew,
                                                            svec);
      mix_kernel<<<N_NODES / 16, 128, 0, stream>>>(svec, thSw, bvec,
                                                   F + (size_t)k * NVEC);
    }
  }

  dec_kernel<<<N_NODES / 16, 32, 0, stream>>>(ucur, decWsw, (float*)d_out);
}
